// GRUMemory_76759655514597
// MI455X (gfx1250) — matrix operand order repaired and checked by reading, not compiled, not run on hardware
//
#include <hip/hip_runtime.h>

#define N_NODES 50000
#define DIM_MEM 512
#define DIM_MSG 1280
#define N_MSG   100000
#define MS_TILES ((N_NODES + 31) / 32)      // 1563 super-tiles of 32 rows
#define NPAD     (MS_TILES * 32)            // 50016 padded rows

typedef __attribute__((ext_vector_type(16))) __bf16 v16bf;
typedef __attribute__((ext_vector_type(8)))  float  v8f;

union FragB16 { v16bf v; unsigned int u[8]; };

__device__ __forceinline__ unsigned short f32_to_bf16(float f) {
    union { float f; unsigned int u; } x; x.f = f;
    unsigned int u = x.u;
    unsigned int r = u + 0x7FFFu + ((u >> 16) & 1u);   // round-to-nearest-even
    return (unsigned short)(r >> 16);
}

#define WMMA_BF16(A, B, C) \
    __builtin_amdgcn_wmma_f32_16x16x32_bf16(false, (A), false, (B), (short)0, (C), false, false)

// ---------------------------------------------------------------------------
// Kernel 1: per-node mean of messages (seg_ids sorted) -> bf16 A-matrix,
// plus message counts (has_msg flags).
// ---------------------------------------------------------------------------
__global__ __launch_bounds__(256)
void gru_seg_mean_bf16(const float* __restrict__ msg,
                       const int*   __restrict__ seg,
                       unsigned short* __restrict__ meansb,
                       int* __restrict__ counts) {
    const int node = blockIdx.x;
    __shared__ int s_start, s_end;
    if (threadIdx.x == 0) {
        int lo = 0, hi = N_MSG;
        while (lo < hi) { int mid = (lo + hi) >> 1; if (seg[mid] < node) lo = mid + 1; else hi = mid; }
        const int start = lo;
        int lo2 = lo; hi = N_MSG;
        while (lo2 < hi) { int mid = (lo2 + hi) >> 1; if (seg[mid] < node + 1) lo2 = mid + 1; else hi = mid; }
        s_start = start; s_end = lo2;
        counts[node] = lo2 - start;
    }
    __syncthreads();
    const int start = s_start, end = s_end;
    const float inv = (end > start) ? (1.0f / (float)(end - start)) : 0.0f;
    for (int d = threadIdx.x; d < DIM_MSG; d += blockDim.x) {
        float s = 0.0f;
        for (int r = start; r < end; ++r) s += msg[(size_t)r * DIM_MSG + d];
        meansb[(size_t)node * DIM_MSG + d] = f32_to_bf16(s * inv);
    }
}

// ---------------------------------------------------------------------------
// Kernel 2: f32 -> bf16 conversion (memory, W_ih, W_hh)
// ---------------------------------------------------------------------------
__global__ __launch_bounds__(256)
void gru_cvt_bf16(const float* __restrict__ src, unsigned short* __restrict__ dst, int n) {
    int i = blockIdx.x * blockDim.x + threadIdx.x;
    const int stride = gridDim.x * blockDim.x;
    for (; i < n; i += stride) dst[i] = f32_to_bf16(src[i]);
}

// ---------------------------------------------------------------------------
// GRU epilogue for one output element (D layout: VGPR e -> M=e+8*half, N=l15)
// ---------------------------------------------------------------------------
__device__ __forceinline__ void gru_elem(float ir, float iz, float in_,
                                         float hr, float hz, float hn,
                                         float brc, float bzc, float binc, float bhnc,
                                         int m, int col,
                                         const float* __restrict__ memory,
                                         const int* __restrict__ counts,
                                         float* __restrict__ out) {
    if (m >= N_NODES) return;
    const float hprev = memory[(size_t)m * DIM_MEM + col];
    const float r = 1.0f / (1.0f + __expf(-(ir + hr + brc)));
    const float z = 1.0f / (1.0f + __expf(-(iz + hz + bzc)));
    const float cand = tanhf(in_ + binc + r * (hn + bhnc));
    const float hnew = (1.0f - z) * cand + z * hprev;
    out[(size_t)m * DIM_MEM + col] = (counts[m] > 0) ? hnew : hprev;
}

// ---------------------------------------------------------------------------
// Kernel 3: fused GRU via WMMA. One wave => 32x16 tile of h_new[N,512]
// (two 16x16 sub-tiles sharing the three gate B-fragments). All 8 waves of
// a block share the same N-tile so B loads hit L0. 12 f32 accumulators.
// ---------------------------------------------------------------------------
__global__ __launch_bounds__(256)
void gru_fused_wmma(const unsigned short* __restrict__ meansb,
                    const unsigned short* __restrict__ memb,
                    const unsigned short* __restrict__ wihb,
                    const unsigned short* __restrict__ whhb,
                    const float* __restrict__ memory,
                    const float* __restrict__ b_ih,
                    const float* __restrict__ b_hh,
                    const int*   __restrict__ counts,
                    float* __restrict__ out) {
    const int lane = threadIdx.x & 31;
    const int wave = threadIdx.x >> 5;
    const int msuper = (blockIdx.x >> 5) * 8 + wave;   // 32-row super tile
    if (msuper >= MS_TILES) return;
    const int j0 = (blockIdx.x & 31) << 4;             // shared across block's waves
    const int m0 = msuper << 5;

    const int half  = lane >> 4;
    const int l15   = lane & 15;
    const int akoff = 16 * half;                       // A-frag: lanes 16..31 hold K+16 block
    const int bkoff = 16 * half;                       // B-frag: lanes 16..31 hold K+16 block

    const unsigned short* Ai0 = meansb + (size_t)(m0 + l15) * DIM_MSG;
    const unsigned short* Ai1 = Ai0 + (size_t)16 * DIM_MSG;
    const unsigned short* Ah0 = memb + (size_t)(m0 + l15) * DIM_MEM;
    const unsigned short* Ah1 = Ah0 + (size_t)16 * DIM_MEM;
    const unsigned short* Bi  = wihb + (size_t)(j0 + l15) * DIM_MSG;   // gate strides -> imm offsets
    const unsigned short* Bh  = whhb + (size_t)(j0 + l15) * DIM_MEM;

    v8f ir0 = {}, ir1 = {}, iz0 = {}, iz1 = {}, in0 = {}, in1 = {};
    v8f hr0 = {}, hr1 = {}, hz0 = {}, hz1 = {}, hn0 = {}, hn1 = {};

    // ---- gi = means @ W_ih.T   (K = 1280: 40 steps x 6 WMMA) ----
    for (int k0 = 0; k0 < DIM_MSG; k0 += 32) {
        FragB16 a0, a1, br, bz, bn;
#pragma unroll
        for (int v = 0; v < 8; ++v) {
            const int ak = k0 + akoff + 2 * v;
            const int bk = k0 + bkoff + 2 * v;
            a0.u[v] = *(const unsigned int*)(Ai0 + ak);
            a1.u[v] = *(const unsigned int*)(Ai1 + ak);
            br.u[v] = *(const unsigned int*)(Bi + bk);
            bz.u[v] = *(const unsigned int*)(Bi + (size_t)DIM_MEM * DIM_MSG + bk);
            bn.u[v] = *(const unsigned int*)(Bi + (size_t)2 * DIM_MEM * DIM_MSG + bk);
        }
        ir0 = WMMA_BF16(a0.v, br.v, ir0);  ir1 = WMMA_BF16(a1.v, br.v, ir1);
        iz0 = WMMA_BF16(a0.v, bz.v, iz0);  iz1 = WMMA_BF16(a1.v, bz.v, iz1);
        in0 = WMMA_BF16(a0.v, bn.v, in0);  in1 = WMMA_BF16(a1.v, bn.v, in1);
    }

    // ---- gh = memory @ W_hh.T  (K = 512: 16 steps x 6 WMMA) ----
    for (int k0 = 0; k0 < DIM_MEM; k0 += 32) {
        FragB16 a0, a1, br, bz, bn;
#pragma unroll
        for (int v = 0; v < 8; ++v) {
            const int ak = k0 + akoff + 2 * v;
            const int bk = k0 + bkoff + 2 * v;
            a0.u[v] = *(const unsigned int*)(Ah0 + ak);
            a1.u[v] = *(const unsigned int*)(Ah1 + ak);
            br.u[v] = *(const unsigned int*)(Bh + bk);
            bz.u[v] = *(const unsigned int*)(Bh + (size_t)DIM_MEM * DIM_MEM + bk);
            bn.u[v] = *(const unsigned int*)(Bh + (size_t)2 * DIM_MEM * DIM_MEM + bk);
        }
        hr0 = WMMA_BF16(a0.v, br.v, hr0);  hr1 = WMMA_BF16(a1.v, br.v, hr1);
        hz0 = WMMA_BF16(a0.v, bz.v, hz0);  hz1 = WMMA_BF16(a1.v, bz.v, hz1);
        hn0 = WMMA_BF16(a0.v, bn.v, hn0);  hn1 = WMMA_BF16(a1.v, bn.v, hn1);
    }

    // ---- fused GRU epilogue ----
    const int col = j0 + l15;
    const float brc  = b_ih[col]               + b_hh[col];
    const float bzc  = b_ih[DIM_MEM + col]     + b_hh[DIM_MEM + col];
    const float binc = b_ih[2 * DIM_MEM + col];
    const float bhnc = b_hh[2 * DIM_MEM + col];

#pragma unroll
    for (int e = 0; e < 8; ++e) {
        const int mA = m0 + e + 8 * half;
        const int mB = mA + 16;
        gru_elem(ir0[e], iz0[e], in0[e], hr0[e], hz0[e], hn0[e],
                 brc, bzc, binc, bhnc, mA, col, memory, counts, out);
        gru_elem(ir1[e], iz1[e], in1[e], hr1[e], hz1[e], hn1[e],
                 brc, bzc, binc, bhnc, mB, col, memory, counts, out);
    }
}

// ---------------------------------------------------------------------------
// Host launcher
// ---------------------------------------------------------------------------
static size_t align256(size_t x) { return (x + 255u) & ~(size_t)255u; }

extern "C" void kernel_launch(void* const* d_in, const int* in_sizes, int n_in,
                              void* d_out, int out_size, void* d_ws, size_t ws_size,
                              hipStream_t stream) {
    const float* msg    = (const float*)d_in[0];
    const int*   seg    = (const int*)  d_in[1];
    const float* memory = (const float*)d_in[2];
    const float* W_ih   = (const float*)d_in[3];
    const float* W_hh   = (const float*)d_in[4];
    const float* b_ih   = (const float*)d_in[5];
    const float* b_hh   = (const float*)d_in[6];
    float* out = (float*)d_out;

    char* ws = (char*)d_ws;
    size_t off = 0;
    unsigned short* meansb = (unsigned short*)(ws + off); off = align256(off + (size_t)NPAD * DIM_MSG * 2);
    unsigned short* memb   = (unsigned short*)(ws + off); off = align256(off + (size_t)NPAD * DIM_MEM * 2);
    unsigned short* wihb   = (unsigned short*)(ws + off); off = align256(off + (size_t)3 * DIM_MEM * DIM_MSG * 2);
    unsigned short* whhb   = (unsigned short*)(ws + off); off = align256(off + (size_t)3 * DIM_MEM * DIM_MEM * 2);
    int*            counts = (int*)(ws + off);            off = align256(off + (size_t)NPAD * 4);
    (void)ws_size; (void)n_in; (void)in_sizes; (void)out_size;

    // Zero the row padding of the bf16 A-matrices (stream-ordered, capture-safe)
    hipMemsetAsync(meansb + (size_t)N_NODES * DIM_MSG, 0,
                   (size_t)(NPAD - N_NODES) * DIM_MSG * 2, stream);
    hipMemsetAsync(memb + (size_t)N_NODES * DIM_MEM, 0,
                   (size_t)(NPAD - N_NODES) * DIM_MEM * 2, stream);

    // 1) segment mean -> bf16 A matrix + counts
    gru_seg_mean_bf16<<<N_NODES, 256, 0, stream>>>(msg, seg, meansb, counts);

    // 2) bf16 conversions of memory and weights
    gru_cvt_bf16<<<2048, 256, 0, stream>>>(memory, memb, N_NODES * DIM_MEM);
    gru_cvt_bf16<<<2048, 256, 0, stream>>>(W_ih, wihb, 3 * DIM_MEM * DIM_MSG);
    gru_cvt_bf16<<<1024, 256, 0, stream>>>(W_hh, whhb, 3 * DIM_MEM * DIM_MEM);

    // 3) fused WMMA GRU: 1563 super-tiles x 32 N-tiles, 8 waves/block,
    //    each block pinned to one N-tile for L0 reuse of B fragments.
    const int blocks = ((MS_TILES + 7) / 8) * 32;      // 196 * 32 = 6272
    gru_fused_wmma<<<blocks, 256, 0, stream>>>(meansb, memb, wihb, whhb,
                                               memory, b_ih, b_hh, counts, out);
}
